// LinearAttention_36472862277841
// MI455X (gfx1250) — matrix operand order repaired and checked by reading, not compiled, not run on hardware
//
#include <hip/hip_runtime.h>
#include <hip/hip_bf16.h>

typedef __attribute__((ext_vector_type(16))) __bf16 v16bf;
typedef __attribute__((ext_vector_type(8)))  __bf16 v8bf;
typedef __attribute__((ext_vector_type(8)))  float  v8f;

#define NTOK   4096   // B*L
#define DMODEL 1024
#define LSEQ   2048
#define NH     16
#define HD     64

// ---------------------------------------------------------------------------
// Fragment loaders (wave32, 16x16x32 bf16 WMMA layouts per CDNA5 ISA)
// A 16x32 (MxK): lane r=lane&15 holds row r; kh=lane>>4 selects K blocks:
//   elems 0..7  -> K = kh*8 + e ; elems 8..15 -> K = 16 + kh*8 + (e-8)
// ---------------------------------------------------------------------------
__device__ __forceinline__ v16bf load_a16(const __bf16* p, int ld, int row0,
                                          int kb, int lane) {
    int r = lane & 15, kh = lane >> 4;
    const __bf16* q = p + (size_t)(row0 + r) * ld + kb;
    v16bf a;
#pragma unroll
    for (int e = 0; e < 8; ++e) a[e] = q[kh * 8 + e];
#pragma unroll
    for (int e = 0; e < 8; ++e) a[8 + e] = q[16 + kh * 8 + e];
    return a;
}

// B 32x16 (KxN) where B[k][n] = S[n0+n][kb+k]  (S row-major NxK, i.e. B = S^T)
// B layout: lane c=lane&15 holds column c; kh=lane>>4: elems 0..7 -> K = kh*8 + e ; elems 8..15 -> K = 16 + kh*8 + (e-8)
__device__ __forceinline__ v16bf load_bt16(const __bf16* p, int ld, int n0,
                                           int kb, int lane) {
    int c = lane & 15, kh = lane >> 4;
    const __bf16* q = p + (size_t)(n0 + c) * ld + kb;
    v16bf b;
#pragma unroll
    for (int e = 0; e < 8; ++e) b[e] = q[kh * 8 + e];
#pragma unroll
    for (int e = 0; e < 8; ++e) b[8 + e] = q[16 + kh * 8 + e];
    return b;
}

__device__ __forceinline__ v8f wmma_bf16(v16bf a, v16bf b, v8f c) {
    return __builtin_amdgcn_wmma_f32_16x16x32_bf16(false, a, false, b,
                                                   (short)0, c, false, false);
}

// ---------------------------------------------------------------------------
// Kernel 0: f32 -> bf16 conversion
// ---------------------------------------------------------------------------
__global__ void __launch_bounds__(256) f2bf_kernel(const float* __restrict__ in,
                                                   __bf16* __restrict__ out,
                                                   int n) {
    int i = blockIdx.x * 256 + threadIdx.x;
    if (i < n) out[i] = (__bf16)in[i];
}

// ---------------------------------------------------------------------------
// Kernel 1: qkv = x @ Wqkv^T + b.
// One wave computes a 32(M) x 64(N) tile (halves weight re-reads vs M=16).
// q,k stored [B,H,L,d] bf16 (k pre-scaled 0.125); v stored TRANSPOSED
// [B,H,d,L] bf16 so the attention kernel gets contiguous B-fragments.
// ---------------------------------------------------------------------------
__global__ void __launch_bounds__(256)
qkv_gemm_kernel(const __bf16* __restrict__ xb, const __bf16* __restrict__ wb,
                const float* __restrict__ bias, __bf16* __restrict__ qo,
                __bf16* __restrict__ ko, __bf16* __restrict__ vt) {
    int lane = threadIdx.x & 31;
    int wid  = blockIdx.x * 8 + (threadIdx.x >> 5);
    int mt = wid / 48;          // 128 M-tiles of 32 tokens
    int nt = wid % 48;          // 48 N-tiles of 64 outputs

    v8f zero = {0.f, 0.f, 0.f, 0.f, 0.f, 0.f, 0.f, 0.f};
    v8f acc[2][4] = {{zero, zero, zero, zero}, {zero, zero, zero, zero}};

    for (int kb = 0; kb < DMODEL; kb += 32) {
        v16bf a0 = load_a16(xb, DMODEL, mt * 32, kb, lane);
        v16bf a1 = load_a16(xb, DMODEL, mt * 32 + 16, kb, lane);
#pragma unroll
        for (int j = 0; j < 4; ++j) {
            v16bf b = load_bt16(wb, DMODEL, nt * 64 + j * 16, kb, lane);
            acc[0][j] = wmma_bf16(a0, b, acc[0][j]);
            acc[1][j] = wmma_bf16(a1, b, acc[1][j]);
        }
    }

    int t = nt / 16, h = nt % 16;
    int c = lane & 15, kh = lane >> 4;

    if (t == 2) {
        // v: transposed store. Lane holds 8 consecutive seq positions for one
        // fixed feature dd -> single 16B vector store per sub-tile.
#pragma unroll
        for (int g = 0; g < 2; ++g) {
            int n0 = mt * 32 + g * 16 + 8 * kh;   // first token row this lane
            int b2 = n0 >> 11, l0 = n0 & (LSEQ - 1);
#pragma unroll
            for (int j = 0; j < 4; ++j) {
                float bv = bias[nt * 64 + j * 16 + c];
                int dd = j * 16 + c;
                v8bf pack;
#pragma unroll
                for (int v = 0; v < 8; ++v)
                    pack[v] = (__bf16)(acc[g][j][v] + bv);
                size_t idx = (((size_t)(b2 * NH + h)) * HD + dd) * LSEQ + l0;
                *(v8bf*)&vt[idx] = pack;
            }
        }
    } else {
        __bf16* dst = (t == 0) ? qo : ko;
        float scale = (t == 1) ? 0.125f : 1.0f;  // k pre-scaled by 1/sqrt(64)
#pragma unroll
        for (int g = 0; g < 2; ++g) {
#pragma unroll
            for (int j = 0; j < 4; ++j) {
                float bv = bias[nt * 64 + j * 16 + c];
                int dd = j * 16 + c;
#pragma unroll
                for (int v = 0; v < 8; ++v) {
                    int n = mt * 32 + g * 16 + v + 8 * kh;
                    int b2 = n >> 11, l = n & (LSEQ - 1);
                    size_t idx =
                        (((size_t)(b2 * NH + h)) * LSEQ + l) * HD + dd;
                    dst[idx] = (__bf16)((acc[g][j][v] + bv) * scale);
                }
            }
        }
    }
}

// ---------------------------------------------------------------------------
// Kernel 2: causal linear attention, flash-style tiles.
// Wave = one (b,h, 32-row q tile) -> 16 WMMAs per 32-key chunk, K/V fragments
// amortized over 32 query rows. Tiles paired (i, 63-i) for triangle balance.
// S bounced through wave-private LDS (32x32 bf16) to re-layout C->A frag.
// V is pre-transposed [B,H,d,L]; y written bf16 token-major [NTOK, DMODEL].
// ---------------------------------------------------------------------------
__global__ void __launch_bounds__(256)
attn_kernel(const __bf16* __restrict__ q, const __bf16* __restrict__ k,
            const __bf16* __restrict__ vt, __bf16* __restrict__ y) {
    __shared__ __bf16 sbuf[8][32 * 32];
    int lane = threadIdx.x & 31;
    int widx = threadIdx.x >> 5;
    int wid  = blockIdx.x * 8 + widx;
    int bh = wid >> 6;                    // (b,h) pair, 32 total
    int i  = wid & 63;
    int qt = (i & 1) ? (63 - (i >> 1)) : (i >> 1);   // balanced pairing
    int b2 = bh >> 4, h = bh & 15;
    const __bf16* qp  = q  + (size_t)bh * LSEQ * HD;
    const __bf16* kp  = k  + (size_t)bh * LSEQ * HD;
    const __bf16* vtp = vt + (size_t)bh * HD * LSEQ;
    int qr0 = qt * 32;

    v16bf aq00 = load_a16(qp, HD, qr0, 0, lane);
    v16bf aq01 = load_a16(qp, HD, qr0, 32, lane);
    v16bf aq10 = load_a16(qp, HD, qr0 + 16, 0, lane);
    v16bf aq11 = load_a16(qp, HD, qr0 + 16, 32, lane);

    v8f zero = {0.f, 0.f, 0.f, 0.f, 0.f, 0.f, 0.f, 0.f};
    v8f acc[2][4] = {{zero, zero, zero, zero}, {zero, zero, zero, zero}};

    int c = lane & 15, kh = lane >> 4;
    __bf16* sw = &sbuf[widx][0];

    for (int ch = 0; ch <= qt; ++ch) {    // last chunk holds the diagonal
        int kc = ch * 32;

        if (ch < qt) {  // prefetch next chunk's K rows / V columns into L2
            __builtin_prefetch(&kp[(size_t)(kc + 32 + c) * HD], 0, 1);
            __builtin_prefetch(&vtp[(size_t)(c * 4 + kh) * LSEQ + kc + 32], 0, 1);
        }

        // S[32q, 32kv] = Q[32,64] @ K^T[64,32]; K frags shared by both
        // 16-row query groups.
        v8f s[2][2] = {{zero, zero}, {zero, zero}};
#pragma unroll
        for (int sub = 0; sub < 2; ++sub) {
            v16bf bk0 = load_bt16(kp, HD, kc + sub * 16, 0, lane);
            v16bf bk1 = load_bt16(kp, HD, kc + sub * 16, 32, lane);
            s[0][sub] = wmma_bf16(aq00, bk0, s[0][sub]);
            s[0][sub] = wmma_bf16(aq01, bk1, s[0][sub]);
            s[1][sub] = wmma_bf16(aq10, bk0, s[1][sub]);
            s[1][sub] = wmma_bf16(aq11, bk1, s[1][sub]);
        }

        if (ch == qt) {                   // causal mask on diagonal chunk
#pragma unroll
            for (int g = 0; g < 2; ++g)
#pragma unroll
                for (int sub = 0; sub < 2; ++sub)
#pragma unroll
                    for (int vv = 0; vv < 8; ++vv) {
                        int qrow = qr0 + g * 16 + vv + 8 * kh;
                        if (kc + sub * 16 + c > qrow) s[g][sub][vv] = 0.f;
                    }
        }

        // re-layout S (C-frag) -> row-major 32x32 bf16 in LDS -> A-frags
#pragma unroll
        for (int g = 0; g < 2; ++g)
#pragma unroll
            for (int vv = 0; vv < 8; ++vv) {
                int row = g * 16 + vv + 8 * kh;
                sw[row * 32 + c]      = (__bf16)s[g][0][vv];
                sw[row * 32 + 16 + c] = (__bf16)s[g][1][vv];
            }
        v16bf as0 = load_a16(sw, 32, 0, 0, lane);
        v16bf as1 = load_a16(sw, 32, 16, 0, lane);

        // y[32,64] += S[32,32] @ V[32,64]; B[k][n] = Vt[f0+n][kc+k]
#pragma unroll
        for (int j = 0; j < 4; ++j) {
            v16bf bv = load_bt16(vtp, LSEQ, j * 16, kc, lane);
            acc[0][j] = wmma_bf16(as0, bv, acc[0][j]);
            acc[1][j] = wmma_bf16(as1, bv, acc[1][j]);
        }
    }

    // store y token-major, feature = h*64 + j*16 + c
#pragma unroll
    for (int g = 0; g < 2; ++g)
#pragma unroll
        for (int j = 0; j < 4; ++j) {
            int f = h * HD + j * 16 + c;
#pragma unroll
            for (int vv = 0; vv < 8; ++vv) {
                int n = b2 * LSEQ + qr0 + g * 16 + vv + 8 * kh;
                y[(size_t)n * DMODEL + f] = (__bf16)acc[g][j][vv];
            }
        }
}

// ---------------------------------------------------------------------------
// Kernel 3: out = y @ out_w^T + out_b  (fp32 result into d_out), M=32 tiles
// ---------------------------------------------------------------------------
__global__ void __launch_bounds__(256)
out_gemm_kernel(const __bf16* __restrict__ yb, const __bf16* __restrict__ wb,
                const float* __restrict__ bias, float* __restrict__ out) {
    int lane = threadIdx.x & 31;
    int wid  = blockIdx.x * 8 + (threadIdx.x >> 5);
    int mt = wid >> 4;          // 128 M-tiles of 32
    int nt = wid & 15;          // 16 N-tiles of 64

    v8f zero = {0.f, 0.f, 0.f, 0.f, 0.f, 0.f, 0.f, 0.f};
    v8f acc[2][4] = {{zero, zero, zero, zero}, {zero, zero, zero, zero}};

    for (int kb = 0; kb < DMODEL; kb += 32) {
        v16bf a0 = load_a16(yb, DMODEL, mt * 32, kb, lane);
        v16bf a1 = load_a16(yb, DMODEL, mt * 32 + 16, kb, lane);
#pragma unroll
        for (int j = 0; j < 4; ++j) {
            v16bf b = load_bt16(wb, DMODEL, nt * 64 + j * 16, kb, lane);
            acc[0][j] = wmma_bf16(a0, b, acc[0][j]);
            acc[1][j] = wmma_bf16(a1, b, acc[1][j]);
        }
    }

    int c = lane & 15, kh = lane >> 4;
#pragma unroll
    for (int g = 0; g < 2; ++g)
#pragma unroll
        for (int j = 0; j < 4; ++j) {
            int o = nt * 64 + j * 16 + c;
            float bv = bias[o];
#pragma unroll
            for (int v = 0; v < 8; ++v) {
                int n = mt * 32 + g * 16 + v + 8 * kh;
                out[(size_t)n * DMODEL + o] = acc[g][j][v] + bv;
            }
        }
}

// ---------------------------------------------------------------------------
extern "C" void kernel_launch(void* const* d_in, const int* in_sizes, int n_in,
                              void* d_out, int out_size, void* d_ws,
                              size_t ws_size, hipStream_t stream) {
    const float* x      = (const float*)d_in[0];   // [2,2048,1024]
    const float* Wqkv_w = (const float*)d_in[1];   // [3072,1024]
    const float* Wqkv_b = (const float*)d_in[2];   // [3072]
    const float* out_w  = (const float*)d_in[3];   // [1024,1024]
    const float* out_b  = (const float*)d_in[4];   // [1024]
    float* out = (float*)d_out;                    // [2,2048,1024]

    char* ws = (char*)d_ws;
    const size_t MB = 1024 * 1024;
    __bf16* xb  = (__bf16*)(ws + 0 * MB);    // 8 MiB  [4096,1024]
    __bf16* wqb = (__bf16*)(ws + 8 * MB);    // 6 MiB  [3072,1024]
    __bf16* owb = (__bf16*)(ws + 14 * MB);   // 2 MiB  [1024,1024]
    __bf16* qb  = (__bf16*)(ws + 16 * MB);   // 8 MiB  [B,H,L,d]
    __bf16* kb  = (__bf16*)(ws + 24 * MB);   // 8 MiB  [B,H,L,d]
    __bf16* vtb = (__bf16*)(ws + 32 * MB);   // 8 MiB  [B,H,d,L] (transposed)
    __bf16* yb  = (__bf16*)(ws + 40 * MB);   // 8 MiB  [4096,1024]

    // convert inputs to bf16
    {
        int n = NTOK * DMODEL;
        f2bf_kernel<<<(n + 255) / 256, 256, 0, stream>>>(x, xb, n);
    }
    {
        int n = 3 * DMODEL * DMODEL;
        f2bf_kernel<<<(n + 255) / 256, 256, 0, stream>>>(Wqkv_w, wqb, n);
    }
    {
        int n = DMODEL * DMODEL;
        f2bf_kernel<<<(n + 255) / 256, 256, 0, stream>>>(out_w, owb, n);
    }

    // QKV projection: 128 M-tiles * 48 N-tiles = 6144 waves / 8 per block
    qkv_gemm_kernel<<<768, 256, 0, stream>>>(xb, wqb, Wqkv_b, qb, kb, vtb);

    // attention: 2*16*64 = 2048 waves / 8 per block
    attn_kernel<<<256, 256, 0, stream>>>(qb, kb, vtb, yb);

    // output projection: 128 * 16 = 2048 waves / 8 per block
    out_gemm_kernel<<<256, 256, 0, stream>>>(yb, owb, out_b, out);
}
